// DecodeYoloV2_14559939133571
// MI455X (gfx1250) — compile-verified, matrix operand order checked by reading, not run on hardware
//
#include <hip/hip_runtime.h>

typedef __attribute__((ext_vector_type(16))) _Float16 v16h;
typedef __attribute__((ext_vector_type(8)))  float    v8f;
typedef __attribute__((ext_vector_type(4)))  unsigned int u32x4;
typedef __attribute__((ext_vector_type(8)))  int      i32x8;
typedef __attribute__((ext_vector_type(4)))  int      i32x4;

#define NUM_CLASSES 20
#define GRID_H 40
#define GRID_W 40
#define PLANE  (GRID_H * GRID_W)      // 1600
#define NBOX   8000                   // 5 anchors * 40 * 40
#define STRIDE_PX 8.0f                // 320 / 40
#define CONF_THR_C 0.5f
#define IOU_THR_C  0.5f
#define EPS_IOU    1e-6f

__device__ __forceinline__ float sigmoidf(float v) {
    return 1.0f / (1.0f + expf(-v));
}

// ---------------------------------------------------------------------------
// Kernel 1: decode — one thread per box. Box flat order matches the
// reference reshape/transpose: p = ((a*40 + gy)*40 + gx).
// ---------------------------------------------------------------------------
__global__ void decode_kernel(const float* __restrict__ x,
                              const float* __restrict__ anchors,
                              float* __restrict__ dec) {
    int p = blockIdx.x * blockDim.x + threadIdx.x;
    if (p >= NBOX) return;
    int a  = p / PLANE;
    int r  = p - a * PLANE;
    int gy = r / GRID_W;
    int gx = r - gy * GRID_W;
    int base = (a * (5 + NUM_CLASSES)) * PLANE + gy * GRID_W + gx;

    __builtin_prefetch(x + base + 5 * PLANE, 0, 3);   // global_prefetch_b8

    float tx   = sigmoidf(x[base]);
    float ty   = sigmoidf(x[base + PLANE]);
    float tw   = x[base + 2 * PLANE];
    float th   = x[base + 3 * PLANE];
    float conf = sigmoidf(x[base + 4 * PLANE]);

    // argmax over classes; sigmoid is monotonic so compare raw logits.
    // strict '>' keeps the FIRST max, matching jnp.argmax.
    float best = -INFINITY;
    int bc = 0;
#pragma unroll
    for (int c = 0; c < NUM_CLASSES; ++c) {
        float v = x[base + (5 + c) * PLANE];
        if (v > best) { best = v; bc = c; }
    }

    float aw = anchors[a * 2 + 0];
    float ah = anchors[a * 2 + 1];
    float bx = (tx + (float)gx) * STRIDE_PX;
    float by = (ty + (float)gy) * STRIDE_PX;
    float bw = expf(tw) * aw * STRIDE_PX;
    float bh = expf(th) * ah * STRIDE_PX;

    float* o = dec + (size_t)p * 6;
    o[0] = bx; o[1] = by; o[2] = bw; o[3] = bh; o[4] = conf; o[5] = (float)bc;
}

// ---------------------------------------------------------------------------
// Kernel 2: stable descending rank via WMMA.
//   rank[i] = #{ j : conf[j] > conf[i] }  +  #{ j < i : conf[j] == conf[i] }
// Expressed as (0/1 indicator matrix) x (ones) using v_wmma_f32_16x16x32_f16,
// with TWO independent accumulator chains so back-to-back WMMAs don't
// serialize on the C-operand RAW. Ranks are exact (0/1 in f16, f32 sums
// < 2^24), so splitting the K-sum is lossless.
//
// The conf gather (every 6th dword of dec) is done by the Tensor Data Mover:
// D# describes a 2D tensor with dim0-stride 6, tile 1x8000, 4B elements,
// compacted into 32 KB of LDS. One wave issues tensor_load_to_lds, waits on
// TENSORcnt, and the workgroup barrier publishes the data to all waves.
// ---------------------------------------------------------------------------
__global__ void __launch_bounds__(256)
rank_kernel(const float* __restrict__ dec, int* __restrict__ order) {
    __shared__ float cf[NBOX];                      // 32 KB conf cache

#if __has_builtin(__builtin_amdgcn_tensor_load_to_lds) && \
    __has_builtin(__builtin_amdgcn_s_wait_tensorcnt)
    if ((threadIdx.x >> 5) == 0) {                  // wave 0 issues the TDM op
        unsigned long long gaddr =
            (unsigned long long)(const void*)(dec + 4); // tile start: conf of box 0
        unsigned int lds_off = (unsigned int)(unsigned long long)(void*)&cf[0];

        u32x4 g0;
        g0[0] = 1u;                                  // count=1, user mode
        g0[1] = lds_off;                             // D#.lds_addr
        g0[2] = (unsigned int)(gaddr & 0xFFFFFFFFull);        // global_addr[31:0]
        g0[3] = (unsigned int)((gaddr >> 32) & 0x01FFFFFFull) // global_addr[56:32]
              | (2u << 30);                          // type = 2 ("image")

        i32x8 g1;
        g1[0] = (int)(2u << 16);                     // wg_mask=0, data_size=2 (4B)
        g1[1] = (int)(6u << 16);                     // tensor_dim0 = 6 (row = 6 f32)
        g1[2] = (int)(8000u << 16);                  // tensor_dim1 = 8000 rows
        g1[3] = (int)(1u << 16);                     // tile_dim0 = 1 element
        g1[4] = (int)8000;                           // tile_dim1 = 8000, tile_dim2 = 0
        g1[5] = (int)6;                              // tensor_dim0_stride = 6
        g1[6] = 0;                                   // stride hi / dim1_stride lo
        g1[7] = 0;                                   // tensor_dim1_stride hi

        i32x4 g2 = {0, 0, 0, 0};                     // unused (<=2D tensor)
        i32x4 g3 = {0, 0, 0, 0};
        i32x8 g4 = {0, 0, 0, 0, 0, 0, 0, 0};         // extra group (6-arg form)

        // This toolchain declares the 6-arg builtin:
        // (u32x4 g0, i32x8 g1, i32x4, i32x4, i32x8, i32 cpol)
        __builtin_amdgcn_tensor_load_to_lds(g0, g1, g2, g3, g4, 0);
        __builtin_amdgcn_s_wait_tensorcnt(0);        // s_wait_tensorcnt 0
    }
#else
    for (int s = threadIdx.x; s < NBOX; s += blockDim.x)
        cf[s] = dec[(size_t)s * 6 + 4];
#endif
    __syncthreads();

    int wave = threadIdx.x >> 5;
    int lane = threadIdx.x & 31;
    int tile = blockIdx.x * (256 / 32) + wave;      // 16-row tile index
    if (tile >= NBOX / 16) return;                  // wave-uniform: EXEC stays full

    int m   = lane & 15;
    int kh8 = (lane >> 4) * 8;
    int i   = tile * 16 + m;
    float ci = cf[i];

    v8f acc0 = {};
    v8f acc1 = {};
    v16h bones;
#pragma unroll
    for (int e = 0; e < 16; ++e) bones[e] = (_Float16)1.0f;

    for (int jb = 0; jb < NBOX; jb += 64) {
        v16h av0, av1;
#pragma unroll
        for (int e = 0; e < 16; ++e) {
            int K  = kh8 + e + ((e < 8) ? 0 : 8);
            int j0 = jb + K;
            int j1 = jb + 32 + K;
            float c0 = cf[j0];
            float c1 = cf[j1];
            bool b0 = (c0 > ci) || ((c0 == ci) && (j0 < i));
            bool b1 = (c1 > ci) || ((c1 == ci) && (j1 < i));
            av0[e] = b0 ? (_Float16)1.0f : (_Float16)0.0f;
            av1[e] = b1 ? (_Float16)1.0f : (_Float16)0.0f;
        }
        acc0 = __builtin_amdgcn_wmma_f32_16x16x32_f16(
            false, av0, false, bones, (short)0, acc0, false, false);
        acc1 = __builtin_amdgcn_wmma_f32_16x16x32_f16(
            false, av1, false, bones, (short)0, acc1, false, false);
    }

    // Column 0 lives in lane 0 (rows 0..7) and lane 16 (rows 8..15).
    if ((lane & 15) == 0) {
        int mb = (lane >> 4) * 8;
#pragma unroll
        for (int r = 0; r < 8; ++r) {
            int row = tile * 16 + mb + r;
            int rk  = (int)(acc0[r] + acc1[r] + 0.5f);
            order[rk] = row;                        // scatter: rank -> box id
        }
    }
}

// ---------------------------------------------------------------------------
// Kernel 3: gather sorted boxes into 224 KB of CDNA5 WGP LDS (320 KB max),
// run the sequential suppression loop with one workgroup, write masked output.
// ---------------------------------------------------------------------------
__global__ void __launch_bounds__(1024)
nms_kernel(const float* __restrict__ dec, const int* __restrict__ order,
           float* __restrict__ out) {
    extern __shared__ unsigned char smem_raw[];
    float* cx   = (float*)smem_raw;
    float* cy   = cx   + NBOX;
    float* bw   = cy   + NBOX;
    float* bh   = bw   + NBOX;
    float* cfv  = bh   + NBOX;
    float* clsv = cfv  + NBOX;
    int*   keep = (int*)(clsv + NBOX);

    int t = threadIdx.x;
    for (int s = t; s < NBOX; s += blockDim.x) {
        int src = order[s];
        const float* d = dec + (size_t)src * 6;
        float a0 = d[0], a1 = d[1], a2 = d[2], a3 = d[3], a4 = d[4], a5 = d[5];
        cx[s] = a0; cy[s] = a1; bw[s] = a2; bh[s] = a3; cfv[s] = a4; clsv[s] = a5;
        keep[s] = (a4 > CONF_THR_C) ? 1 : 0;
    }
    __syncthreads();

    for (int i = 0; i < NBOX; ++i) {
        if (keep[i]) {                              // uniform read after barrier
            float xi1 = cx[i] - bw[i] * 0.5f, yi1 = cy[i] - bh[i] * 0.5f;
            float xi2 = cx[i] + bw[i] * 0.5f, yi2 = cy[i] + bh[i] * 0.5f;
            float areai = fabsf((xi2 - xi1) * (yi2 - yi1));
            float ci = clsv[i];
            for (int j = i + 1 + t; j < NBOX; j += blockDim.x) {
                if (!keep[j] || clsv[j] != ci) continue;
                float xj1 = cx[j] - bw[j] * 0.5f, yj1 = cy[j] - bh[j] * 0.5f;
                float xj2 = cx[j] + bw[j] * 0.5f, yj2 = cy[j] + bh[j] * 0.5f;
                float iw = fminf(xi2, xj2) - fmaxf(xi1, xj1); iw = fmaxf(iw, 0.0f);
                float ih = fminf(yi2, yj2) - fmaxf(yi1, yj1); ih = fmaxf(ih, 0.0f);
                float inter = iw * ih;
                float areaj = fabsf((xj2 - xj1) * (yj2 - yj1));
                float iou = inter / (areai + areaj - inter + EPS_IOU);
                if (iou >= IOU_THR_C) keep[j] = 0;  // disjoint j per thread
            }
        }
        __syncthreads();
    }

    for (int s = t; s < NBOX; s += blockDim.x) {
        float f = keep[s] ? 1.0f : 0.0f;
        float* o = out + (size_t)s * 6;
        o[0] = cx[s]  * f; o[1] = cy[s]  * f; o[2] = bw[s]   * f;
        o[3] = bh[s]  * f; o[4] = cfv[s] * f; o[5] = clsv[s] * f;
    }
}

// ---------------------------------------------------------------------------
extern "C" void kernel_launch(void* const* d_in, const int* in_sizes, int n_in,
                              void* d_out, int out_size, void* d_ws, size_t ws_size,
                              hipStream_t stream) {
    (void)in_sizes; (void)n_in; (void)out_size; (void)ws_size;
    const float* x       = (const float*)d_in[0];   // (1,125,40,40) f32
    const float* anchors = (const float*)d_in[1];   // (5,2) f32
    float* dec  = (float*)d_ws;                     // 8000*6 f32 = 192 KB
    int*   ord  = (int*)(dec + (size_t)NBOX * 6);   // 8000 i32  =  32 KB

    decode_kernel<<<(NBOX + 255) / 256, 256, 0, stream>>>(x, anchors, dec);

    int tiles  = NBOX / 16;                         // 500 row tiles
    int blocks = (tiles + 7) / 8;                   // 8 waves per block
    rank_kernel<<<blocks, 256, 0, stream>>>(dec, ord);

    size_t smem = (size_t)NBOX * 4u * 7u;           // 224000 B of 320 KB LDS
    nms_kernel<<<1, 1024, smem, stream>>>(dec, ord, (float*)d_out);
}
